// Decoder_88175678587287
// MI455X (gfx1250) — matrix operand order repaired and checked by reading, not compiled, not run on hardware
//
#include <hip/hip_runtime.h>
#include <hip/hip_bf16.h>
#include <stdint.h>

#define BZ   256
#define TT   128
#define LL   64
#define EENC 512
#define DD   512
#define EMBD 128
#define AA   129
#define ATTH 20

typedef __attribute__((ext_vector_type(16))) __bf16 v16bf;
typedef __attribute__((ext_vector_type(8)))  float  v8f;

union Frag16 {
  v16bf v;
  unsigned short u[16];
  uint4 q[2];
};

__device__ __forceinline__ unsigned short f2bf(float f) {
  uint32_t u = __float_as_uint(f);
  uint32_t r = (u + 0x7FFFu + ((u >> 16) & 1u)) >> 16;   // round-to-nearest-even
  return (unsigned short)r;
}
__device__ __forceinline__ float bf2f(unsigned short h) {
  return __uint_as_float(((uint32_t)h) << 16);
}
__device__ __forceinline__ float sigm(float x) { return 1.0f / (1.0f + __expf(-x)); }

// A fragment: 16x32 bf16, activations row-major [M][K].
// lanes 0-15: M=lane, K = kb+{0..15}; lanes 16-31: M=lane-16, K = kb+{16..31}
__device__ __forceinline__ v16bf loadA(const unsigned short* __restrict__ A,
                                       int mBase, int ldk, int kb, int lane) {
  int hf = lane >> 4, r = lane & 15;
  const unsigned short* row = A + (size_t)(mBase + r) * ldk;
  Frag16 f;
  f.q[0] = *(const uint4*)(row + kb + hf * 16);
  f.q[1] = *(const uint4*)(row + kb + hf * 16 + 8);
  return f.v;
}
// B fragment: 32x16 bf16, weight row-major [N][K] (torch layout).
// lane: N = nBase + lane&15; K = kb + (lane>>4)*16 .. +15 (contiguous)
__device__ __forceinline__ v16bf loadB(const unsigned short* __restrict__ W,
                                       int nBase, int ldk, int kb, int lane) {
  const unsigned short* row = W + (size_t)(nBase + (lane & 15)) * ldk;
  int k = kb + (lane >> 4) * 16;
  Frag16 f;
  f.q[0] = *(const uint4*)(row + k);
  f.q[1] = *(const uint4*)(row + k + 8);
  return f.v;
}
__device__ __forceinline__ v16bf loadBg(const unsigned short* __restrict__ W,
                                        int nBase, int ldk, int kb, int lane, int nMax) {
  int n = nBase + (lane & 15);
  Frag16 f;
  if (n < nMax) {
    const unsigned short* row = W + (size_t)n * ldk;
    int k = kb + (lane >> 4) * 16;
    f.q[0] = *(const uint4*)(row + k);
    f.q[1] = *(const uint4*)(row + k + 8);
  } else {
    f.q[0] = make_uint4(0, 0, 0, 0);
    f.q[1] = make_uint4(0, 0, 0, 0);
  }
  return f.v;
}

#define WMMA_BF16(a, b, c) \
  __builtin_amdgcn_wmma_f32_16x16x32_bf16(false, (a), false, (b), (short)0, (c), false, false)

// ---------------- prep kernels ----------------

__global__ void k_cvt(const float* __restrict__ src, unsigned short* __restrict__ dst, int n) {
  int i = blockIdx.x * blockDim.x + threadIdx.x;
  if (i < n) dst[i] = f2bf(src[i]);
}

// embseq[(t*BZ+b)*EMBD + e] = (t==0) ? 0 : emb_W[actions[b][t-1]][e]
__global__ void k_embed(const int* __restrict__ act, const float* __restrict__ embW,
                        unsigned short* __restrict__ out) {
  int i = blockIdx.x * blockDim.x + threadIdx.x;
  if (i >= TT * BZ * EMBD) return;
  int e = i & (EMBD - 1);
  int bt = i >> 7;            // EMBD = 2^7
  int b = bt & (BZ - 1);
  int t = bt >> 8;            // BZ = 2^8
  float v = 0.0f;
  if (t > 0) {
    int a = act[b * TT + (t - 1)];
    v = embW[a * EMBD + e];
  }
  out[i] = f2bf(v);
}

// S1[(b*LL+l)*ATTH + j] = sent[b,l,:] . Wa1[j,:EENC] + ba1[j]  (time-invariant attention part)
__global__ void k_s1(const float* __restrict__ sent, const float* __restrict__ Wa1,
                     const float* __restrict__ ba1, float* __restrict__ S1) {
  int i = blockIdx.x * blockDim.x + threadIdx.x;
  if (i >= BZ * LL * ATTH) return;
  int j = i % ATTH;
  int bl = i / ATTH;
  const float* sr = sent + (size_t)bl * EENC;
  const float* wr = Wa1 + (size_t)j * (EENC + DD);
  float s = ba1[j];
  for (int k = 0; k < EENC; ++k) s += sr[k] * wr[k];
  S1[i] = s;
}

__global__ void k_init(const float* __restrict__ ench, const float* __restrict__ encc,
                       unsigned short* h0, unsigned short* h1, unsigned short* h2,
                       float* c0, float* c1, float* c2, unsigned short* ctxbf) {
  int i = blockIdx.x * blockDim.x + threadIdx.x;
  if (i >= BZ * DD) return;
  unsigned short hb = f2bf(ench[i]);
  h0[i] = hb; h1[i] = hb; h2[i] = hb;
  float c = encc[i];
  c0[i] = c; c1[i] = c; c2[i] = c;
  ctxbf[i] = 0;
}

// ---------------- per-step kernels ----------------

// att_t = [e_tm1, ctx] @ Wc.T + bc   (M=256, N=512, K=640) -> bf16 X1
__global__ void k_combine(const unsigned short* __restrict__ embt,
                          const unsigned short* __restrict__ ctxbf,
                          const unsigned short* __restrict__ WcB,
                          const float* __restrict__ bc,
                          unsigned short* __restrict__ X1) {
  int wave = (blockIdx.x * blockDim.x + threadIdx.x) >> 5;
  int lane = threadIdx.x & 31;
  int mBase = (wave >> 5) * 16;   // 16 m-tiles
  int nBase = (wave & 31) * 16;   // 32 n-tiles
  v8f acc = {};
  for (int kb = 0; kb < EMBD; kb += 32) {
    v16bf a = loadA(embt, mBase, EMBD, kb, lane);
    acc = WMMA_BF16(a, loadB(WcB, nBase, EMBD + EENC, kb, lane), acc);
  }
  for (int kb = 0; kb < EENC; kb += 32) {
    v16bf a = loadA(ctxbf, mBase, EENC, kb, lane);
    acc = WMMA_BF16(a, loadB(WcB, nBase, EMBD + EENC, EMBD + kb, lane), acc);
  }
  int n = nBase + (lane & 15);
  int hf = lane >> 4;
  float bias = bc[n];
#pragma unroll
  for (int v = 0; v < 8; ++v) {
    int m = mBase + hf * 8 + v;
    X1[(size_t)m * DD + n] = f2bf(acc[v] + bias);
  }
}

// fused LSTM cell: one wave owns a (16m x 16d) tile and all 4 gate columns.
__global__ void k_lstm(const unsigned short* __restrict__ X,
                       const unsigned short* __restrict__ H,
                       const unsigned short* __restrict__ Wih,
                       const unsigned short* __restrict__ Whh,
                       const float* __restrict__ bih, const float* __restrict__ bhh,
                       float* __restrict__ cstate, unsigned short* __restrict__ hout,
                       unsigned short* __restrict__ hw) {
  int wave = (blockIdx.x * blockDim.x + threadIdx.x) >> 5;
  int lane = threadIdx.x & 31;
  int mBase = (wave >> 5) * 16;   // 16 m-tiles
  int nBase = (wave & 31) * 16;   // 32 d-tiles
  v8f a0 = {}, a1 = {}, a2 = {}, a3 = {};
  for (int kb = 0; kb < DD; kb += 32) {
    v16bf a = loadA(X, mBase, DD, kb, lane);        // A shared across 4 gates
    a0 = WMMA_BF16(a, loadB(Wih, 0 * DD + nBase, DD, kb, lane), a0);
    a1 = WMMA_BF16(a, loadB(Wih, 1 * DD + nBase, DD, kb, lane), a1);
    a2 = WMMA_BF16(a, loadB(Wih, 2 * DD + nBase, DD, kb, lane), a2);
    a3 = WMMA_BF16(a, loadB(Wih, 3 * DD + nBase, DD, kb, lane), a3);
  }
  for (int kb = 0; kb < DD; kb += 32) {
    v16bf a = loadA(H, mBase, DD, kb, lane);
    a0 = WMMA_BF16(a, loadB(Whh, 0 * DD + nBase, DD, kb, lane), a0);
    a1 = WMMA_BF16(a, loadB(Whh, 1 * DD + nBase, DD, kb, lane), a1);
    a2 = WMMA_BF16(a, loadB(Whh, 2 * DD + nBase, DD, kb, lane), a2);
    a3 = WMMA_BF16(a, loadB(Whh, 3 * DD + nBase, DD, kb, lane), a3);
  }
  int n = nBase + (lane & 15);
  int hf = lane >> 4;
  float b0 = bih[0 * DD + n] + bhh[0 * DD + n];
  float b1 = bih[1 * DD + n] + bhh[1 * DD + n];
  float b2 = bih[2 * DD + n] + bhh[2 * DD + n];
  float b3 = bih[3 * DD + n] + bhh[3 * DD + n];
#pragma unroll
  for (int v = 0; v < 8; ++v) {
    int m = mBase + hf * 8 + v;
    size_t idx = (size_t)m * DD + n;
    float gi = a0[v] + b0, gf = a1[v] + b1, gg = a2[v] + b2, go = a3[v] + b3;
    float c = cstate[idx];
    float cn = sigm(gf) * c + sigm(gi) * tanhf(gg);
    float hn = sigm(go) * tanhf(cn);
    cstate[idx] = cn;
    unsigned short hb = f2bf(hn);
    hout[idx] = hb;
    if (hw) hw[(size_t)m * 1024 + n] = hb;
  }
}

// attention: one wave per batch row. Uses precomputed S1; only h2-part computed here.
__global__ void k_attn(const unsigned short* __restrict__ h2, const float* __restrict__ S1,
                       const float* __restrict__ Wa1, const float* __restrict__ Wa2,
                       const float* __restrict__ ba2, const float* __restrict__ sent,
                       unsigned short* __restrict__ ctxbf, unsigned short* __restrict__ hw) {
  __shared__ float sm[8][96];          // per wave: [0..19]=Hh, [20..83]=softmax weights
  int wv = threadIdx.x >> 5, lane = threadIdx.x & 31;
  int b = blockIdx.x * 8 + wv;
  if (lane < ATTH) {
    const float* wr = Wa1 + (size_t)lane * (EENC + DD) + EENC;
    const unsigned short* hr = h2 + (size_t)b * DD;
    float s = 0.f;
    for (int k = 0; k < DD; ++k) s += bf2f(hr[k]) * wr[k];
    sm[wv][lane] = s;
  }
  __syncthreads();
  float scv[2];
#pragma unroll
  for (int i = 0; i < 2; ++i) {
    int l = lane + i * 32;
    const float* s1 = S1 + ((size_t)b * LL + l) * ATTH;
    float v = ba2[0];
    for (int j = 0; j < ATTH; ++j) {
      float x = s1[j] + sm[wv][j];
      v += (x > 0.f ? x : 0.f) * Wa2[j];
    }
    scv[i] = tanhf(v);
  }
  float mx = fmaxf(scv[0], scv[1]);
  for (int o = 16; o > 0; o >>= 1) mx = fmaxf(mx, __shfl_xor(mx, o, 32));
  float e0 = __expf(scv[0] - mx), e1 = __expf(scv[1] - mx);
  float ssum = e0 + e1;
  for (int o = 16; o > 0; o >>= 1) ssum += __shfl_xor(ssum, o, 32);
  float inv = 1.f / ssum;
  sm[wv][ATTH + lane] = e0 * inv;
  sm[wv][ATTH + 32 + lane] = e1 * inv;
  __syncthreads();
  const float* sb = sent + (size_t)b * LL * EENC;
  for (int e = lane; e < EENC; e += 32) {
    float a = 0.f;
    for (int l = 0; l < LL; ++l) a += sm[wv][ATTH + l] * sb[(size_t)l * EENC + e];
    unsigned short cb = f2bf(a);
    ctxbf[(size_t)b * EENC + e] = cb;
    hw[(size_t)b * 1024 + DD + e] = cb;     // second half of [h2, ctx]
  }
}

// ---------------- batched action head ----------------

// Y = relu(HWchunk[4096,1024] @ Wl1.T + bl1) -> bf16
__global__ void k_head1(const unsigned short* __restrict__ HW,
                        const unsigned short* __restrict__ Wl1B,
                        const float* __restrict__ bl1, unsigned short* __restrict__ Y) {
  int wave = (blockIdx.x * blockDim.x + threadIdx.x) >> 5;
  int lane = threadIdx.x & 31;
  int mBase = (wave >> 5) * 16;   // 256 m-tiles
  int nBase = (wave & 31) * 16;   // 32 n-tiles
  v8f acc = {};
  for (int kb = 0; kb < 1024; kb += 32) {
    acc = WMMA_BF16(loadA(HW, mBase, 1024, kb, lane),
                    loadB(Wl1B, nBase, 1024, kb, lane), acc);
  }
  int n = nBase + (lane & 15);
  int hf = lane >> 4;
  float bias = bl1[n];
#pragma unroll
  for (int v = 0; v < 8; ++v) {
    float x = acc[v] + bias;
    Y[(size_t)(mBase + hf * 8 + v) * DD + n] = f2bf(x > 0.f ? x : 0.f);
  }
}

// logits = Y[4096,512] @ Wl2.T + bl2 -> fp32 d_out with (t,b)->(b,t) row remap
__global__ void k_head2(const unsigned short* __restrict__ Y,
                        const unsigned short* __restrict__ Wl2B,
                        const float* __restrict__ bl2, float* __restrict__ out, int rowBase) {
  int wave = (blockIdx.x * blockDim.x + threadIdx.x) >> 5;
  int lane = threadIdx.x & 31;
  int mBase = (wave / 9) * 16;    // 256 m-tiles
  int nBase = (wave % 9) * 16;    // 9 n-tiles (144 >= 129)
  v8f acc = {};
  for (int kb = 0; kb < DD; kb += 32) {
    acc = WMMA_BF16(loadA(Y, mBase, DD, kb, lane),
                    loadBg(Wl2B, nBase, DD, kb, lane, AA), acc);
  }
  int n = nBase + (lane & 15);
  int hf = lane >> 4;
  if (n < AA) {
    float bias = bl2[n];
#pragma unroll
    for (int v = 0; v < 8; ++v) {
      int R = rowBase + mBase + hf * 8 + v;   // R = t*BZ + b
      int t = R >> 8;
      int b = R & (BZ - 1);
      out[((size_t)b * TT + t) * AA + n] = acc[v] + bias;
    }
  }
}

// ---------------- host orchestration ----------------

extern "C" void kernel_launch(void* const* d_in, const int* in_sizes, int n_in,
                              void* d_out, int out_size, void* d_ws, size_t ws_size,
                              hipStream_t stream) {
  (void)in_sizes; (void)n_in; (void)out_size; (void)ws_size;
  const int*   act  = (const int*)  d_in[0];
  const float* sent = (const float*)d_in[1];
  const float* ench = (const float*)d_in[2];
  const float* encc = (const float*)d_in[3];
  const float* embW = (const float*)d_in[4];
  const float* Wc   = (const float*)d_in[5];
  const float* bc   = (const float*)d_in[6];
  const float* Wih1 = (const float*)d_in[7],  *Whh1 = (const float*)d_in[8];
  const float* bih1 = (const float*)d_in[9],  *bhh1 = (const float*)d_in[10];
  const float* Wih2 = (const float*)d_in[11], *Whh2 = (const float*)d_in[12];
  const float* bih2 = (const float*)d_in[13], *bhh2 = (const float*)d_in[14];
  const float* Wih3 = (const float*)d_in[15], *Whh3 = (const float*)d_in[16];
  const float* bih3 = (const float*)d_in[17], *bhh3 = (const float*)d_in[18];
  const float* Wa1  = (const float*)d_in[19], *ba1  = (const float*)d_in[20];
  const float* Wa2  = (const float*)d_in[21], *ba2  = (const float*)d_in[22];
  const float* Wl1  = (const float*)d_in[23], *bl1  = (const float*)d_in[24];
  const float* Wl2  = (const float*)d_in[25], *bl2  = (const float*)d_in[26];
  float* out = (float*)d_out;

  char* p = (char*)d_ws;
  auto alloc = [&](size_t bytes) -> char* {
    char* r = p; p += (bytes + 255) & ~(size_t)255; return r;
  };
  unsigned short* embseq = (unsigned short*)alloc((size_t)TT * BZ * EMBD * 2);
  float*          S1     = (float*)         alloc((size_t)BZ * LL * ATTH * 4);
  unsigned short* WcB    = (unsigned short*)alloc((size_t)DD * (EMBD + EENC) * 2);
  unsigned short* Wih1B  = (unsigned short*)alloc((size_t)4 * DD * DD * 2);
  unsigned short* Whh1B  = (unsigned short*)alloc((size_t)4 * DD * DD * 2);
  unsigned short* Wih2B  = (unsigned short*)alloc((size_t)4 * DD * DD * 2);
  unsigned short* Whh2B  = (unsigned short*)alloc((size_t)4 * DD * DD * 2);
  unsigned short* Wih3B  = (unsigned short*)alloc((size_t)4 * DD * DD * 2);
  unsigned short* Whh3B  = (unsigned short*)alloc((size_t)4 * DD * DD * 2);
  unsigned short* Wl1B   = (unsigned short*)alloc((size_t)DD * (DD + EENC) * 2);
  unsigned short* Wl2B   = (unsigned short*)alloc((size_t)AA * DD * 2);
  unsigned short* X1     = (unsigned short*)alloc((size_t)BZ * DD * 2);
  unsigned short* h0a    = (unsigned short*)alloc((size_t)BZ * DD * 2);
  unsigned short* h0b    = (unsigned short*)alloc((size_t)BZ * DD * 2);
  unsigned short* h1a    = (unsigned short*)alloc((size_t)BZ * DD * 2);
  unsigned short* h1b    = (unsigned short*)alloc((size_t)BZ * DD * 2);
  unsigned short* h2a    = (unsigned short*)alloc((size_t)BZ * DD * 2);
  unsigned short* h2b    = (unsigned short*)alloc((size_t)BZ * DD * 2);
  float*          c0     = (float*)         alloc((size_t)BZ * DD * 4);
  float*          c1     = (float*)         alloc((size_t)BZ * DD * 4);
  float*          c2     = (float*)         alloc((size_t)BZ * DD * 4);
  unsigned short* ctxbf  = (unsigned short*)alloc((size_t)BZ * EENC * 2);
  unsigned short* HW     = (unsigned short*)alloc((size_t)TT * BZ * 1024 * 2);
  unsigned short* Y      = (unsigned short*)alloc((size_t)4096 * DD * 2);

  auto cvt = [&](const float* s, unsigned short* d, int n) {
    k_cvt<<<(n + 255) / 256, 256, 0, stream>>>(s, d, n);
  };
  cvt(Wc, WcB, DD * (EMBD + EENC));
  cvt(Wih1, Wih1B, 4 * DD * DD); cvt(Whh1, Whh1B, 4 * DD * DD);
  cvt(Wih2, Wih2B, 4 * DD * DD); cvt(Whh2, Whh2B, 4 * DD * DD);
  cvt(Wih3, Wih3B, 4 * DD * DD); cvt(Whh3, Whh3B, 4 * DD * DD);
  cvt(Wl1, Wl1B, DD * (DD + EENC));
  cvt(Wl2, Wl2B, AA * DD);
  k_embed<<<(TT * BZ * EMBD + 255) / 256, 256, 0, stream>>>(act, embW, embseq);
  k_s1<<<(BZ * LL * ATTH + 255) / 256, 256, 0, stream>>>(sent, Wa1, ba1, S1);
  k_init<<<(BZ * DD + 255) / 256, 256, 0, stream>>>(ench, encc, h0a, h1a, h2a, c0, c1, c2, ctxbf);

  for (int t = 0; t < TT; ++t) {
    const unsigned short* h0i = (t & 1) ? h0b : h0a;  unsigned short* h0o = (t & 1) ? h0a : h0b;
    const unsigned short* h1i = (t & 1) ? h1b : h1a;  unsigned short* h1o = (t & 1) ? h1a : h1b;
    const unsigned short* h2i = (t & 1) ? h2b : h2a;  unsigned short* h2o = (t & 1) ? h2a : h2b;
    unsigned short* hwslice = HW + (size_t)t * BZ * 1024;
    k_combine<<<64, 256, 0, stream>>>(embseq + (size_t)t * BZ * EMBD, ctxbf, WcB, bc, X1);
    k_lstm<<<64, 256, 0, stream>>>(X1,  h0i, Wih1B, Whh1B, bih1, bhh1, c0, h0o, (unsigned short*)nullptr);
    k_lstm<<<64, 256, 0, stream>>>(h0o, h1i, Wih2B, Whh2B, bih2, bhh2, c1, h1o, (unsigned short*)nullptr);
    k_lstm<<<64, 256, 0, stream>>>(h1o, h2i, Wih3B, Whh3B, bih3, bhh3, c2, h2o, hwslice);
    k_attn<<<32, 256, 0, stream>>>(h2o, S1, Wa1, Wa2, ba2, sent, ctxbf, hwslice);
  }

  for (int c = 0; c < 8; ++c) {
    int rowBase = c * 4096;
    k_head1<<<1024, 256, 0, stream>>>(HW + (size_t)rowBase * 1024, Wl1B, bl1, Y);
    k_head2<<<288, 256, 0, stream>>>(Y, Wl2B, bl2, out, rowBase);
  }
}
